// Gate_81209241633270
// MI455X (gfx1250) — matrix operand order repaired and checked by reading, not compiled, not run on hardware
//
#include <hip/hip_runtime.h>
#include <hip/hip_bf16.h>
#include <math.h>

// ---------------- problem constants (from reference) ----------------
constexpr int N_TOKENS = 16384;
constexpr int DIM      = 7168;
constexpr int N_EXP    = 256;
constexpr int N_GROUPS = 8;
constexpr int GRP_SZ   = N_EXP / N_GROUPS;   // 32
constexpr int TOPK_G   = 4;
constexpr int TOPK     = 8;
constexpr float ROUTE_SCALE = 2.5f;

constexpr int MT   = 32;     // tokens per block (2 row tiles of 16)
constexpr int LSTR = 265;    // LDS score row stride (265 mod 64 = 9, conflict-free phase-1 reads)

typedef __attribute__((ext_vector_type(8)))  float  v8f;
typedef __attribute__((ext_vector_type(16))) __bf16 v16bf;

// -------------------------------------------------------------------
// Kernel 1: split fp32 weight into bf16 hi/lo pair (bf16x3 emulation)
// -------------------------------------------------------------------
__global__ void prep_w_kernel(const float* __restrict__ w,
                              __bf16* __restrict__ whi,
                              __bf16* __restrict__ wlo,
                              int n) {
    int i = blockIdx.x * blockDim.x + threadIdx.x;
    if (i < n) {
        float f  = w[i];
        __bf16 h = (__bf16)f;
        whi[i] = h;
        wlo[i] = (__bf16)(f - (float)h);
    }
}

// -------------------------------------------------------------------
// Kernel 2: fused router GEMM (bf16x3 WMMA) + sigmoid + bias + grouped
// top-k gate. One block = 32 tokens x 256 experts.
// -------------------------------------------------------------------
__global__ __launch_bounds__(256, 1)
void moe_gate_kernel(const float* __restrict__ x,
                     const __bf16* __restrict__ whi,
                     const __bf16* __restrict__ wlo,
                     const float* __restrict__ bias,
                     float* __restrict__ out) {
    __shared__ float sc[MT * LSTR];              // biased sigmoid scores
    __shared__ float gtv[MT][N_GROUPS][TOPK];    // per-group sorted top-8 values (desc)
    __shared__ int   gti[MT][N_GROUPS][TOPK];    // matching expert indices
    __shared__ float gs2[MT][N_GROUPS];          // per-group top-2 sum

    const int tid    = threadIdx.x;
    const int wave   = tid >> 5;
    const int lane   = tid & 31;
    const int l16    = lane & 15;
    const int hi16   = (lane >> 4) & 1;        // lane-half select
    const int kbA    = hi16 * 8;               // half the A fragment K base (A loads at 2*kbA)
    const int kbB    = hi16 * 16;              // B fragment K base (contiguous 16)
    const int rowT   = wave >> 2;              // 0..1  : which 16-token row tile
    const int colG   = wave & 3;               // 0..3  : which group of 4 expert tiles

    const int token  = blockIdx.x * MT + rowT * 16 + l16;
    const float* xrow = x + (size_t)token * DIM;

    // per-lane expert pointers for the 4 column tiles this wave owns
    const __bf16* bh[4];
    const __bf16* bl[4];
    float bv[4];
#pragma unroll
    for (int ci = 0; ci < 4; ++ci) {
        int e = (colG * 4 + ci) * 16 + l16;
        size_t off = (size_t)e * DIM;
        bh[ci] = whi + off;
        bl[ci] = wlo + off;
        bv[ci] = bias[e];
    }

    v8f acc[4];
#pragma unroll
    for (int ci = 0; ci < 4; ++ci) acc[ci] = (v8f)0.0f;

    for (int k0 = 0; k0 < DIM; k0 += 32) {
        // prefetch next x chunk (global_prefetch_b8; speculative, OOB dropped)
        __builtin_prefetch(xrow + k0 + 128, 0, 0);

        // ---- A fragment: load 16 fp32, split into bf16 hi/lo ----
        // A order used here (contiguous, not ISA table order): lane half selects K base {0,16};
        // element j -> K = 2*kbA + j  => contiguous 16 floats (matches B).
        v8f a0 = *(const v8f*)(xrow + k0 + 2 * kbA);
        v8f a1 = *(const v8f*)(xrow + k0 + 2 * kbA + 8);
        v16bf Ah, Al;
#pragma unroll
        for (int i = 0; i < 8; ++i) {
            float f0 = a0[i];
            __bf16 h0 = (__bf16)f0;
            Ah[i]     = h0;
            Al[i]     = (__bf16)(f0 - (float)h0);
            float f1 = a1[i];
            __bf16 h1 = (__bf16)f1;
            Ah[i + 8] = h1;
            Al[i + 8] = (__bf16)(f1 - (float)h1);
        }

        // ---- B fragments + 3 WMMAs per column tile (hi*hi + hi*lo + lo*hi) ----
#pragma unroll
        for (int ci = 0; ci < 4; ++ci) {
            // B order used here (same as A, not ISA table order): lane half holds contiguous K 0-15 / 16-31
            v16bf Bh = *(const v16bf*)(bh[ci] + k0 + kbB);
            v16bf Bl = *(const v16bf*)(bl[ci] + k0 + kbB);
            acc[ci] = __builtin_amdgcn_wmma_f32_16x16x32_bf16(
                false, Ah, false, Bh, (short)0, acc[ci], false, false);
            acc[ci] = __builtin_amdgcn_wmma_f32_16x16x32_bf16(
                false, Ah, false, Bl, (short)0, acc[ci], false, false);
            acc[ci] = __builtin_amdgcn_wmma_f32_16x16x32_bf16(
                false, Al, false, Bh, (short)0, acc[ci], false, false);
        }
    }

    // ---- epilogue: sigmoid + bias -> LDS score tile ----
    // C layout: lanes 0-15: N=lane, M=v ; lanes 16-31: N=lane-16, M=v+8
#pragma unroll
    for (int ci = 0; ci < 4; ++ci) {
        int col = (colG * 4 + ci) * 16 + l16;
#pragma unroll
        for (int v = 0; v < 8; ++v) {
            int row = rowT * 16 + v + hi16 * 8;
            float d = acc[ci][v];
            float s = 1.0f / (1.0f + __expf(-d)) + bv[ci];
            sc[row * LSTR + col] = s;
        }
    }
    __syncthreads();

    // ================= phase 2a: per-group sorted top-8 ==================
    // 256 threads = 32 tokens x 8 groups. Stable descending insertion:
    // equal values keep the earlier (lower expert index) entry first,
    // matching lax.top_k tie-breaking.
    {
        const int tok = tid >> 3;
        const int g   = tid & 7;
        const float* s = &sc[tok * LSTR + g * GRP_SZ];

        float tv[TOPK];
        int   ti[TOPK];
#pragma unroll
        for (int r = 0; r < TOPK; ++r) { tv[r] = -3.4e38f; ti[r] = 0x7fffffff; }

#pragma unroll 1
        for (int j = 0; j < GRP_SZ; ++j) {
            float v = s[j];
            if (v > tv[TOPK - 1]) {            // strict >: ties keep existing lower index
                int p = TOPK - 1;
                while (p > 0 && v > tv[p - 1]) {
                    tv[p] = tv[p - 1];
                    ti[p] = ti[p - 1];
                    --p;
                }
                tv[p] = v;
                ti[p] = g * GRP_SZ + j;
            }
        }
#pragma unroll
        for (int r = 0; r < TOPK; ++r) {
            gtv[tok][g][r] = tv[r];
            gti[tok][g][r] = ti[r];
        }
        gs2[tok][g] = tv[0] + tv[1];           // sum of top-2 = group score
    }
    __syncthreads();

    // ================= phase 2b: group top-4 + 4-way merge ===============
    if (tid < MT) {
        const int tok = tid;

        // top-4 groups (strict >, ascending scan == lax.top_k tie-break)
        unsigned gmask = 0;
        int selg[TOPK_G];
#pragma unroll 1
        for (int j = 0; j < TOPK_G; ++j) {
            float b = -3.4e38f; int bi = 0;
#pragma unroll
            for (int g = 0; g < N_GROUPS; ++g) {
                if ((gmask >> g) & 1u) continue;
                float v = gs2[tok][g];
                if (v > b) { b = v; bi = g; }
            }
            gmask |= 1u << bi;
            selg[j] = bi;
        }

        // merge the 4 selected groups' sorted lists -> global top-8.
        // Tie-break: larger value first; equal values -> lower expert index.
        int   p[TOPK_G] = {0, 0, 0, 0};
        int   idxs[TOPK];
        float wts[TOPK];
        float wsum = 0.0f;
        bool  fallback = false;
#pragma unroll 1
        for (int j = 0; j < TOPK; ++j) {
            float b = -3.4e38f; int bi = 0x7fffffff; int bm = 0;
#pragma unroll
            for (int m = 0; m < TOPK_G; ++m) {
                if (p[m] < TOPK) {
                    float v  = gtv[tok][selg[m]][p[m]];
                    int   ix = gti[tok][selg[m]][p[m]];
                    if (v > b || (v == b && ix < bi)) { b = v; bi = ix; bm = m; }
                }
            }
            if (b <= 0.0f) { fallback = true; break; }  // masked 0.0 could win: exact path
            ++p[bm];
            idxs[j] = bi;
            wts[j]  = b;       // biased score, per reference quirk
            wsum   += b;
        }

        if (fallback) {
            // Exact reference semantics incl. masked zeros (rare path).
            const float* s = &sc[tok * LSTR];
            unsigned chosen[N_EXP / 32] = {0, 0, 0, 0, 0, 0, 0, 0};
            wsum = 0.0f;
#pragma unroll 1
            for (int j = 0; j < TOPK; ++j) {
                float b = -3.4e38f; int bi = 0;
#pragma unroll 1
                for (int e = 0; e < N_EXP; ++e) {
                    if ((chosen[e >> 5] >> (e & 31)) & 1u) continue;
                    float v = ((gmask >> (e >> 5)) & 1u) ? s[e] : 0.0f;
                    if (v > b) { b = v; bi = e; }
                }
                chosen[bi >> 5] |= 1u << (bi & 31);
                idxs[j] = bi;
                float w = s[bi];
                wts[j]  = w;
                wsum   += w;
            }
        }

        float inv = ROUTE_SCALE / wsum;
        int gt = blockIdx.x * MT + tok;
#pragma unroll
        for (int j = 0; j < TOPK; ++j) {
            out[(size_t)gt * TOPK + j] = wts[j] * inv;
            // indices follow weights in the flat-concatenated tuple output
            out[(size_t)N_TOKENS * TOPK + (size_t)gt * TOPK + j] = (float)idxs[j];
        }
    }
}

// -------------------------------------------------------------------
extern "C" void kernel_launch(void* const* d_in, const int* in_sizes, int n_in,
                              void* d_out, int out_size, void* d_ws, size_t ws_size,
                              hipStream_t stream) {
    const float* x    = (const float*)d_in[0];   // [16384, 7168]
    const float* w    = (const float*)d_in[1];   // [256, 7168]
    const float* bias = (const float*)d_in[2];   // [256]
    float* out        = (float*)d_out;           // [16384*8 weights | 16384*8 indices]

    const int nW = N_EXP * DIM;                  // 1,835,008 elems
    __bf16* whi = (__bf16*)d_ws;                 // 3.67 MB
    __bf16* wlo = whi + nW;                      // 3.67 MB  (total 7.34 MB of d_ws)

    prep_w_kernel<<<(nW + 255) / 256, 256, 0, stream>>>(w, whi, wlo, nW);

    moe_gate_kernel<<<N_TOKENS / MT, 256, 0, stream>>>(x, whi, wlo, bias, out);
}
